// MultiHeadedAttentionDynamicSpan_75436805587621
// MI455X (gfx1250) — matrix operand order repaired and checked by reading, not compiled, not run on hardware
//
#include <hip/hip_runtime.h>
#include <hip/hip_bf16.h>
#include <math.h>

typedef __attribute__((ext_vector_type(16))) _Float16 v16h;
typedef __attribute__((ext_vector_type(8)))  float    v8f;
typedef __attribute__((ext_vector_type(4)))  unsigned int tdm_u32x4;
typedef __attribute__((ext_vector_type(8)))  int          tdm_i32x8;
typedef __attribute__((ext_vector_type(4)))  int          tdm_i32x4;

#define NFEAT  512
#define NHEAD  8
#define DK     64
#define TLEN   2048
#define BATCH  4
#define MAXSPAN 50
#define WT     80   // score tile columns (covers 16 + 50 - 1 = 65, padded to mult of 16)
#define WTP    96   // P columns padded to multiple of 32 for the K-loop

__device__ __forceinline__ int iclamp(int x, int lo, int hi) {
  return x < lo ? lo : (x > hi ? hi : x);
}

// A-fragment for V_WMMA_F32_16X16X32_F16 (16x32 f16), contiguous k order:
// lane L: row = L%16, half = L/16; VGPR j holds K = 2j+16*half (pair).
__device__ __forceinline__ v16h load_a_frag(const _Float16* rowptr, int kbase, int hf) {
  v16h f;
#pragma unroll
  for (int j = 0; j < 8; ++j) {
    int k = kbase + 2 * j + 16 * hf;
    f[2 * j]     = rowptr[k];
    f[2 * j + 1] = rowptr[k + 1];
  }
  return f;
}

// ---------------------------------------------------------------------------
// f32 -> f16 conversion (grid-stride)
// ---------------------------------------------------------------------------
__global__ void f32_to_f16_kernel(const float* __restrict__ in,
                                  _Float16* __restrict__ out, long n) {
  long i = (long)blockIdx.x * blockDim.x + threadIdx.x;
  long stride = (long)gridDim.x * blockDim.x;
  for (; i < n; i += stride) out[i] = (_Float16)in[i];
}

// ---------------------------------------------------------------------------
// WMMA GEMM: C[M,N] = A[M,K] * W[N,K]^T + bias[N],  K == NFEAT
// 128 threads (4 waves). The 64xK weight panel is staged into LDS once per
// block via the Tensor Data Mover (TENSOR_LOAD_TO_LDS + s_wait_tensorcnt),
// then shared by all 4 waves; each wave computes a 16x64 output tile.
// ---------------------------------------------------------------------------
__global__ void __launch_bounds__(128)
gemm64_wmma(const _Float16* __restrict__ A, const _Float16* __restrict__ W,
            const float* __restrict__ bias, _Float16* __restrict__ out16,
            float* __restrict__ out32, int M, int N) {
  __shared__ _Float16 sB[64 * NFEAT];          // 64 KB weight panel

  const int tid  = threadIdx.x;
  const int lane = tid & 31;
  const int wave = tid >> 5;
  const int lr = lane & 15;
  const int hf = lane >> 4;
  const int m0 = blockIdx.y * 64 + wave * 16;
  const int n0 = blockIdx.x * 64;

  // ---- stage W[n0 .. n0+63][0 .. K) into LDS -------------------------------
#if __has_builtin(__builtin_amdgcn_tensor_load_to_lds) && \
    __has_builtin(__builtin_amdgcn_s_wait_tensorcnt)
  if (wave == 0) {
    unsigned long long ga = (unsigned long long)(const void*)(W + (long)n0 * NFEAT);
    unsigned lds_addr = (unsigned)(unsigned long long)(const void*)sB; // addr[31:0] = LDS offset
    tdm_u32x4 g0;
    g0.x = 1u;                                      // count=1, user-mode descriptor
    g0.y = lds_addr;                                // LDS byte address
    g0.z = (unsigned)(ga & 0xFFFFFFFFu);            // global_addr[31:0]
    g0.w = (unsigned)((ga >> 32) & 0x1FFFFFFu) | (2u << 30);  // addr[56:32] | type=2
    tdm_i32x8 g1;
    g1[0] = (1 << 16);                              // data_size = 2 bytes
    g1[1] = (int)((NFEAT & 0xFFFF) << 16);          // tensor_dim0[15:0]
    g1[2] = (int)((NFEAT >> 16) | (64 << 16));      // tensor_dim0[31:16] | tensor_dim1[15:0]
    g1[3] = (int)(NFEAT << 16);                     // tensor_dim1[31:16]=0 | tile_dim0
    g1[4] = 64;                                     // tile_dim1 = 64, tile_dim2 = 0
    g1[5] = NFEAT;                                  // tensor_dim0_stride[31:0]
    g1[6] = 0;                                      // stride0[47:32] | stride1[15:0]
    g1[7] = 0;
    tdm_i32x4 g2 = (tdm_i32x4){0, 0, 0, 0};        // 2D tile: groups 2/3 unused
    tdm_i32x4 g3 = (tdm_i32x4){0, 0, 0, 0};
    tdm_i32x8 g4 = (tdm_i32x8){0, 0, 0, 0, 0, 0, 0, 0};
    __builtin_amdgcn_tensor_load_to_lds(g0, g1, g2, g3, g4, 0);
    __builtin_amdgcn_s_wait_tensorcnt(0);
  }
#else
  // fallback: cooperative copy, 128 threads x 32 x 8 halfs
  for (int i = tid; i < 64 * NFEAT / 8; i += 128) {
    *(((unsigned long long*)sB) + i * 2)     = *(((const unsigned long long*)(W + (long)n0 * NFEAT)) + i * 2);
    *(((unsigned long long*)sB) + i * 2 + 1) = *(((const unsigned long long*)(W + (long)n0 * NFEAT)) + i * 2 + 1);
  }
#endif
  __syncthreads();

  v8f acc[4];
#pragma unroll
  for (int j = 0; j < 4; ++j) acc[j] = (v8f){0.f, 0.f, 0.f, 0.f, 0.f, 0.f, 0.f, 0.f};

  const _Float16* arow = A + (long)(m0 + lr) * NFEAT;
  for (int kb = 0; kb < NFEAT; kb += 32) {
    v16h a = load_a_frag(arow, kb, hf);
#pragma unroll
    for (int j = 0; j < 4; ++j) {
      // B[k,n] = W[n,k]; lane holds column n = n0+16j+lr, elements k contiguous.
      v16h b = *(const v16h*)(sB + (long)(16 * j + lr) * NFEAT + kb + 16 * hf);
      acc[j] = __builtin_amdgcn_wmma_f32_16x16x32_f16(false, a, false, b,
                                                      (short)0, acc[j], false, false);
    }
  }

#pragma unroll
  for (int j = 0; j < 4; ++j) {
    int n = n0 + 16 * j + lr;
    float bsv = bias[n];
#pragma unroll
    for (int r = 0; r < 8; ++r) {
      int m = m0 + r + 8 * hf;                 // C/D layout: lanes 16-31 hold M+8
      float v = acc[j][r] + bsv;
      if (out16) out16[(long)m * N + n] = (_Float16)v;
      if (out32) out32[(long)m * N + n] = v;
    }
  }
}

// ---------------------------------------------------------------------------
// Span head: span[b,h,t] = 50*sigmoid(q_lin[b,t] . Wd[h] + bd[h]); atomic max
// ---------------------------------------------------------------------------
__global__ void span_kernel(const _Float16* __restrict__ Qp,
                            const float* __restrict__ Wd,
                            const float* __restrict__ bd,
                            float* __restrict__ span_arr,
                            int* __restrict__ maxbits) {
  long idx = (long)blockIdx.x * blockDim.x + threadIdx.x;   // BATCH*TLEN*NHEAD
  int h = (int)(idx & (NHEAD - 1));
  long bt = idx >> 3;
  int b = (int)(bt / TLEN);
  int t = (int)(bt % TLEN);
  const _Float16* q = Qp + bt * NFEAT;
  const float* wd = Wd + (long)h * NFEAT;
  float acc = bd[h];
  for (int k = 0; k < NFEAT; ++k) acc += (float)q[k] * wd[k];
  float s = (float)MAXSPAN / (1.f + __expf(-acc));
  span_arr[((long)(b * NHEAD + h)) * TLEN + t] = s;
  atomicMax(maxbits, __float_as_int(s));  // spans > 0 -> int-bit compare is monotonic
}

__global__ void init_kernel(int* maxbits) { *maxbits = 0; }

__global__ void finalize_w_kernel(const int* __restrict__ maxbits,
                                  int* __restrict__ winfo) {
  float ms = __int_as_float(*maxbits);
  int w = (int)ceilf(ms * (float)MAXSPAN) + 2;   // + SPAN_RAMP
  w = w < 0 ? 0 : (w > MAXSPAN ? MAXSPAN : w);
  winfo[0] = w;
  winfo[1] = w >> 1;                              // pad0 = int(w*0.5)
}

// ---------------------------------------------------------------------------
// Banded attention: one wave per (b, h, 16-row tile of t).
// scores = Q(16x64) x K^T(64x80) via WMMA; softmax + span mask in LDS;
// out     = P(16x96) x V(96x64) via WMMA.
// ---------------------------------------------------------------------------
__global__ void __launch_bounds__(32)
attn_wmma(const _Float16* __restrict__ Qp, const _Float16* __restrict__ Kp,
          const _Float16* __restrict__ Vp, const float* __restrict__ span_arr,
          const int* __restrict__ mask, const int* __restrict__ winfo,
          _Float16* __restrict__ X) {
  __shared__ float    sS[16 * WT];
  __shared__ _Float16 sP[16 * WTP];

  const int lane = threadIdx.x;
  const int lr = lane & 15;
  const int hf = lane >> 4;
  const int t0 = blockIdx.x * 16;
  const int h  = blockIdx.y;
  const int b  = blockIdx.z;
  const int w    = winfo[0];
  const int pad0 = winfo[1];
  const int r0 = t0 - pad0;
  const long rowbase = ((long)b * TLEN) * NFEAT + (long)h * DK;

  // ---- scores: 5 column tiles x 2 K-steps = 10 WMMAs -----------------------
  const _Float16* qrow = Qp + rowbase + (long)(t0 + lr) * NFEAT;
  v16h aq0 = load_a_frag(qrow, 0, hf);
  v16h aq1 = load_a_frag(qrow, 32, hf);

  v8f sacc[5];
#pragma unroll
  for (int nt = 0; nt < 5; ++nt) {
    int c = nt * 16 + lr;
    int kt = iclamp(r0 + c, 0, TLEN - 1);        // clamped; masked later
    const _Float16* krow = Kp + rowbase + (long)kt * NFEAT;
    v16h b0 = *(const v16h*)(krow + 16 * hf);        // B[k=d, n=c], d contiguous
    v16h b1 = *(const v16h*)(krow + 32 + 16 * hf);
    v8f acc = (v8f){0.f, 0.f, 0.f, 0.f, 0.f, 0.f, 0.f, 0.f};
    acc = __builtin_amdgcn_wmma_f32_16x16x32_f16(false, aq0, false, b0, (short)0, acc, false, false);
    acc = __builtin_amdgcn_wmma_f32_16x16x32_f16(false, aq1, false, b1, (short)0, acc, false, false);
    sacc[nt] = acc;
  }
#pragma unroll
  for (int nt = 0; nt < 5; ++nt)
#pragma unroll
    for (int r = 0; r < 8; ++r)
      sS[(r + 8 * hf) * WT + nt * 16 + lr] = sacc[nt][r];
  __syncthreads();

  // ---- softmax + span mask + renormalize, one row per lane (lanes 0-15) ----
  if (lane < 16) {
    int m = lane;
    float span = span_arr[((long)(b * NHEAD + h)) * TLEN + (t0 + m)];
    float mx = -3.402823466e38f;
    for (int c = 0; c < WT; ++c) {
      int j = c - m;
      int kt = r0 + c;
      bool valid = (j >= 0) && (j < w) && (kt >= 0) && (kt < TLEN) &&
                   (mask[(long)b * TLEN + kt] != 0);
      float v = valid ? sS[m * WT + c] * 0.125f : -3.402823466e38f;  // /sqrt(64)
      sS[m * WT + c] = v;
      mx = fmaxf(mx, v);
    }
    float sum = 0.f;
    for (int c = 0; c < WT; ++c) {
      float v = sS[m * WT + c];
      float e = (v <= -3.0e38f) ? 0.f : __expf(v - mx);
      sS[m * WT + c] = e;
      sum += e;
    }
    float inv = 1.f / sum;
    float sum2 = 0.f;
    for (int c = 0; c < WT; ++c) {
      int j = c - m;
      float a = sS[m * WT + c] * inv;
      float sm = 0.f;
      if (j >= 0 && j < w) {
        float tm = (float)(j - (w - 1));          // linspace(1-50,0,50)[-w:]
        sm = fminf(1.f, fmaxf(0.f, (tm + span * (float)MAXSPAN) * 0.5f + 1.f));
      }
      a *= sm;
      sS[m * WT + c] = a;
      sum2 += a;
    }
    float inv2 = 1.f / (sum2 + 1e-8f);
    for (int c = 0; c < WT; ++c)  sP[m * WTP + c] = (_Float16)(sS[m * WT + c] * inv2);
    for (int c = WT; c < WTP; ++c) sP[m * WTP + c] = (_Float16)0.f;
  }
  __syncthreads();

  // ---- P x V: 3 K-steps x 4 d-tiles = 12 WMMAs -----------------------------
  v8f o[4];
#pragma unroll
  for (int nt = 0; nt < 4; ++nt) o[nt] = (v8f){0.f, 0.f, 0.f, 0.f, 0.f, 0.f, 0.f, 0.f};

#pragma unroll
  for (int kb = 0; kb < WTP; kb += 32) {
    v16h ap;                                      // A frag of P from LDS
#pragma unroll
    for (int j = 0; j < 8; ++j) {
      int k = kb + 2 * j + 16 * hf;
      ap[2 * j]     = sP[lr * WTP + k];
      ap[2 * j + 1] = sP[lr * WTP + k + 1];
    }
#pragma unroll
    for (int nt = 0; nt < 4; ++nt) {
      int d = nt * 16 + lr;
      v16h bv;                                    // B[k,n] = V[r0+k, d]; rows strided
#pragma unroll
      for (int e = 0; e < 16; ++e) {
        int k = kb + 16 * hf + e;
        int kt = iclamp(r0 + k, 0, TLEN - 1);     // P==0 outside band -> clamp is safe
        bv[e] = Vp[rowbase + (long)kt * NFEAT + d];
      }
      o[nt] = __builtin_amdgcn_wmma_f32_16x16x32_f16(false, ap, false, bv,
                                                     (short)0, o[nt], false, false);
    }
  }

#pragma unroll
  for (int nt = 0; nt < 4; ++nt) {
    int d = nt * 16 + lr;
#pragma unroll
    for (int r = 0; r < 8; ++r) {
      int m = r + 8 * hf;
      X[rowbase + (long)(t0 + m) * NFEAT + d] = (_Float16)o[nt][r];
    }
  }
}

// ---------------------------------------------------------------------------
// Host-side launcher
// ---------------------------------------------------------------------------
extern "C" void kernel_launch(void* const* d_in, const int* in_sizes, int n_in,
                              void* d_out, int out_size, void* d_ws, size_t ws_size,
                              hipStream_t stream) {
  const float* query  = (const float*)d_in[0];
  const float* key_in = (const float*)d_in[1];
  const float* value  = (const float*)d_in[2];
  const int*   mask   = (const int*)d_in[3];
  const float* Wq = (const float*)d_in[4];
  const float* bq = (const float*)d_in[5];
  const float* Wk = (const float*)d_in[6];
  const float* bk = (const float*)d_in[7];
  const float* Wv = (const float*)d_in[8];
  const float* bv = (const float*)d_in[9];
  const float* Wo = (const float*)d_in[10];
  const float* bo = (const float*)d_in[11];
  const float* Wd = (const float*)d_in[12];
  const float* bd = (const float*)d_in[13];

  const long MTOK = (long)BATCH * TLEN;           // 8192 rows
  const size_t ACT = (size_t)MTOK * NFEAT * 2;    // f16 activation buffer
  const size_t WB  = (size_t)NFEAT * NFEAT * 2;   // f16 weight buffer

  char* ws = (char*)d_ws;
  size_t off = 0;
  _Float16* q16  = (_Float16*)(ws + off); off += ACT;
  _Float16* k16  = (_Float16*)(ws + off); off += ACT;
  _Float16* v16  = (_Float16*)(ws + off); off += ACT;
  _Float16* Wq16 = (_Float16*)(ws + off); off += WB;
  _Float16* Wk16 = (_Float16*)(ws + off); off += WB;
  _Float16* Wv16 = (_Float16*)(ws + off); off += WB;
  _Float16* Wo16 = (_Float16*)(ws + off); off += WB;
  _Float16* Qp   = (_Float16*)(ws + off); off += ACT;
  _Float16* Kp   = (_Float16*)(ws + off); off += ACT;
  _Float16* Vp   = (_Float16*)(ws + off); off += ACT;
  _Float16* Xb   = (_Float16*)(ws + off); off += ACT;
  float* span_arr = (float*)(ws + off); off += (size_t)BATCH * NHEAD * TLEN * 4;
  int* scal = (int*)(ws + off); off += 256;       // [0]=maxbits, [8]=w, [9]=pad0
  int* maxbits = scal;
  int* winfo   = scal + 8;

  // 1) f32 -> f16 conversions
  f32_to_f16_kernel<<<2048, 256, 0, stream>>>(query,  q16, MTOK * NFEAT);
  f32_to_f16_kernel<<<2048, 256, 0, stream>>>(key_in, k16, MTOK * NFEAT);
  f32_to_f16_kernel<<<2048, 256, 0, stream>>>(value,  v16, MTOK * NFEAT);
  f32_to_f16_kernel<<<256, 256, 0, stream>>>(Wq, Wq16, (long)NFEAT * NFEAT);
  f32_to_f16_kernel<<<256, 256, 0, stream>>>(Wk, Wk16, (long)NFEAT * NFEAT);
  f32_to_f16_kernel<<<256, 256, 0, stream>>>(Wv, Wv16, (long)NFEAT * NFEAT);
  f32_to_f16_kernel<<<256, 256, 0, stream>>>(Wo, Wo16, (long)NFEAT * NFEAT);

  // 2) Q/K/V projections (TDM-staged WMMA GEMM, f16 outputs)
  dim3 ggrid(NFEAT / 64, (unsigned)(MTOK / 64));
  gemm64_wmma<<<ggrid, 128, 0, stream>>>(q16, Wq16, bq, Qp, nullptr, (int)MTOK, NFEAT);
  gemm64_wmma<<<ggrid, 128, 0, stream>>>(k16, Wk16, bk, Kp, nullptr, (int)MTOK, NFEAT);
  gemm64_wmma<<<ggrid, 128, 0, stream>>>(v16, Wv16, bv, Vp, nullptr, (int)MTOK, NFEAT);

  // 3) span head + window size reduction (on-device, deterministic)
  init_kernel<<<1, 1, 0, stream>>>(maxbits);
  span_kernel<<<(BATCH * TLEN * NHEAD) / 256, 256, 0, stream>>>(Qp, Wd, bd, span_arr, maxbits);
  finalize_w_kernel<<<1, 1, 0, stream>>>(maxbits, winfo);

  // 4) banded attention (WMMA scores + WMMA P*V)
  dim3 agrid(TLEN / 16, NHEAD, BATCH);
  attn_wmma<<<agrid, 32, 0, stream>>>(Qp, Kp, Vp, span_arr, mask, winfo, Xb);

  // 5) output projection -> d_out (f32)
  gemm64_wmma<<<ggrid, 128, 0, stream>>>(Xb, Wo16, bo, nullptr, (float*)d_out,
                                         (int)MTOK, NFEAT);
}
